// ScaledDotProductAttention_21251498181425
// MI455X (gfx1250) — compile-verified, matrix operand order checked by reading, not run on hardware
//
#include <hip/hip_runtime.h>
#include <hip/hip_bf16.h>

typedef __attribute__((ext_vector_type(16))) _Float16 v16h;
typedef __attribute__((ext_vector_type(8)))  _Float16 v8h;
typedef __attribute__((ext_vector_type(8)))  float    v8f;
typedef __attribute__((ext_vector_type(4)))  float    v4f;

#define SEQ 2048
#define DIM 64
#define VSTR 40   // padded t-stride (halves) for transposed V staging in LDS

// Build a 16x32 f16 A/B fragment from a row-major fp32 row.
// Per CDNA5 16-bit A layout: lane L (hi = L>>4) holds K = 8*hi..8*hi+7 and 16+8*hi..16+8*hi+7.
__device__ __forceinline__ v16h frag_from_f32(const float* __restrict__ row, int d0) {
    v4f p0 = *(const v4f*)(row + d0);
    v4f p1 = *(const v4f*)(row + d0 + 4);
    v4f p2 = *(const v4f*)(row + d0 + 16);
    v4f p3 = *(const v4f*)(row + d0 + 20);
    v16h f;
    f[0]  = (_Float16)p0.x; f[1]  = (_Float16)p0.y; f[2]  = (_Float16)p0.z; f[3]  = (_Float16)p0.w;
    f[4]  = (_Float16)p1.x; f[5]  = (_Float16)p1.y; f[6]  = (_Float16)p1.z; f[7]  = (_Float16)p1.w;
    f[8]  = (_Float16)p2.x; f[9]  = (_Float16)p2.y; f[10] = (_Float16)p2.z; f[11] = (_Float16)p2.w;
    f[12] = (_Float16)p3.x; f[13] = (_Float16)p3.y; f[14] = (_Float16)p3.z; f[15] = (_Float16)p3.w;
    return f;
}

__global__ __launch_bounds__(256)
void sdpa_fused_kernel(const float* __restrict__ q,
                       const float* __restrict__ k,
                       const float* __restrict__ v,
                       float* __restrict__ out,
                       float* __restrict__ attn) {
    extern __shared__ char smem[];
    float*    scores  = (float*)smem;                 // 16 x 2048 f32  (131072 B)
    float*    out_lds = scores + 16 * SEQ;            // 16 x 64  f32   (4096 B)
    float*    red     = out_lds + 16 * DIM;           // 16 x 16  f32   (1024 B)
    float*    rowmax  = red + 256;                    // 16 f32
    float*    rowinv  = rowmax + 16;                  // 16 f32
    _Float16* vlds    = (_Float16*)(rowinv + 16);     // 2 x 64 x VSTR f16 (10240 B)

    const int tid  = threadIdx.x;
    const int lane = tid & 31;
    const int w    = tid >> 5;       // wave 0..7
    const int hi   = lane >> 4;      // half-wave select (K sub-range)
    const int lr   = lane & 15;      // row (A) / col (B) within tile
    const int b    = blockIdx.y;
    const int rowbase = blockIdx.x * 16;

    // ---- Q fragments (shared across all column tiles of this wave) ----
    const float* qrow = q + ((size_t)b * SEQ + rowbase + lr) * DIM;
    const v16h A0 = frag_from_f32(qrow, 8 * hi);
    const v16h A1 = frag_from_f32(qrow, 32 + 8 * hi);

    // ---- S = Q K^T * (1/8): each wave does 16 of 128 column tiles ----
    for (int it = 0; it < 16; ++it) {
        const int ct = w + it * 8;
        const int t  = ct * 16 + lr;
        const float* krow = k + ((size_t)b * SEQ + t) * DIM;
        // prefetch next iteration's K row (next tile is +128 rows)
        if (it < 15) __builtin_prefetch(krow + 128 * DIM, 0, 3);
        v16h B0 = frag_from_f32(krow, 8 * hi);
        v16h B1 = frag_from_f32(krow, 32 + 8 * hi);
        v8f c = {};
        c = __builtin_amdgcn_wmma_f32_16x16x32_f16(false, A0, false, B0, (short)0, c, false, false);
        c = __builtin_amdgcn_wmma_f32_16x16x32_f16(false, A1, false, B1, (short)0, c, false, false);
        #pragma unroll
        for (int vv = 0; vv < 8; ++vv)   // C layout: row = vv + 8*hi, col = 16*ct + lr
            scores[(vv + 8 * hi) * SEQ + ct * 16 + lr] = c[vv] * 0.125f;
    }
    __syncthreads();

    // ---- exact softmax stats per row (16 threads per row) ----
    const int r   = tid >> 4;
    const int sub = tid & 15;
    float m = -3.0e38f;
    for (int c = sub; c < SEQ; c += 16) m = fmaxf(m, scores[r * SEQ + c]);
    red[r * 16 + sub] = m;
    __syncthreads();
    if (sub == 0) {
        float mm = red[r * 16];
        for (int j = 1; j < 16; ++j) mm = fmaxf(mm, red[r * 16 + j]);
        rowmax[r] = mm;
    }
    __syncthreads();
    const float mr = rowmax[r];
    float sum = 0.0f;
    for (int c = sub; c < SEQ; c += 16) {
        float e = __expf(scores[r * SEQ + c] - mr);
        scores[r * SEQ + c] = e;   // keep UNNORMALIZED exp in LDS; fold 1/sum into outputs
        sum += e;
    }
    red[r * 16 + sub] = sum;
    __syncthreads();
    if (sub == 0) {
        float ss = 0.0f;
        for (int j = 0; j < 16; ++j) ss += red[r * 16 + j];
        rowinv[r] = 1.0f / ss;
    }
    __syncthreads();

    // ---- attn output: normalize on the fly, NON-TEMPORAL coalesced b128 stores ----
    // (256MB/call stream > 192MB L2: NT keeps K/V L2-resident for the other WGs)
    for (int p4 = tid; p4 < 16 * SEQ / 4; p4 += 256) {
        const int pos = p4 * 4;
        const int rr = pos >> 11;          // /2048
        const int cc = pos & (SEQ - 1);
        const float inv = rowinv[rr];
        v4f val = *(const v4f*)(scores + rr * SEQ + cc);
        val *= inv;
        __builtin_nontemporal_store(val, (v4f*)(attn + ((size_t)b * SEQ + rowbase + rr) * SEQ + cc));
    }
    __syncthreads();

    // ---- O = P V (with P = exp, unnormalized): 4 N-tiles x 2 K-halves across 8 waves ----
    const int ntile = w & 3;
    const int khalf = w >> 2;
    v8f acc = {};
    for (int i = 0; i < 32; ++i) {
        // stage V rows [32i,32i+32) and [1024+32i, 1024+32i+32) as vlds[h][d][t] (f16)
        for (int e = tid; e < 4096; e += 256) {
            const int h   = e >> 11;
            const int rem = e & 2047;
            const int tl  = rem >> 6;
            const int dd  = rem & 63;
            const float val = v[((size_t)b * SEQ + h * 1024 + i * 32 + tl) * DIM + dd];
            vlds[(h * 64 + dd) * VSTR + tl] = (_Float16)val;
        }
        __syncthreads();
        // A fragment from exp-scores in LDS (row-major, contiguous per lane)
        const float* prow = scores + lr * SEQ + khalf * 1024 + i * 32;
        v16h Ap = frag_from_f32(prow, 8 * hi);
        // B fragment: column d fixed per lane, K contiguous thanks to transpose staging
        const _Float16* vp = vlds + ((size_t)(khalf * 64 + ntile * 16 + lr)) * VSTR + 8 * hi;
        v8h blo = *(const v8h*)vp;
        v8h bhi = *(const v8h*)(vp + 16);
        v16h Bp = __builtin_shufflevector(blo, bhi, 0,1,2,3,4,5,6,7,8,9,10,11,12,13,14,15);
        acc = __builtin_amdgcn_wmma_f32_16x16x32_f16(false, Ap, false, Bp, (short)0, acc, false, false);
        __syncthreads();
    }

    // ---- combine the two K-halves per N-tile in LDS (store then add; no pre-zero needed) ----
    if (khalf == 0) {
        #pragma unroll
        for (int vv = 0; vv < 8; ++vv)
            out_lds[(vv + 8 * hi) * DIM + ntile * 16 + lr] = acc[vv];
    }
    __syncthreads();
    if (khalf == 1) {
        #pragma unroll
        for (int vv = 0; vv < 8; ++vv)
            out_lds[(vv + 8 * hi) * DIM + ntile * 16 + lr] += acc[vv];
    }
    __syncthreads();

    // ---- coalesced output store, folding in the softmax normalization ----
    {
        const int rr = tid >> 4;
        const int d4 = (tid & 15) * 4;
        v4f val = *(const v4f*)(out_lds + rr * DIM + d4);
        val *= rowinv[rr];
        *(v4f*)(out + ((size_t)b * SEQ + rowbase + rr) * DIM + d4) = val;
    }
}

extern "C" void kernel_launch(void* const* d_in, const int* in_sizes, int n_in,
                              void* d_out, int out_size, void* d_ws, size_t ws_size,
                              hipStream_t stream) {
    (void)in_sizes; (void)n_in; (void)out_size; (void)d_ws; (void)ws_size;
    const float* q = (const float*)d_in[0];
    const float* k = (const float*)d_in[1];
    const float* v = (const float*)d_in[2];
    float* out  = (float*)d_out;
    float* attn = out + (size_t)16 * SEQ * DIM;   // tuple: (output, attn) concatenated

    const size_t smem = 16 * SEQ * 4       // scores
                      + 16 * DIM * 4       // out accumulation
                      + 256 * 4 + 32 * 4   // reductions
                      + 2 * 64 * VSTR * 2; // V staging
    dim3 grid(SEQ / 16, 16);   // 128 row-blocks x 16 batches
    hipLaunchKernelGGL(sdpa_fused_kernel, grid, dim3(256), smem, stream,
                       q, k, v, out, attn);
}
